// HexConv_4964982194840
// MI455X (gfx1250) — matrix operand order repaired and checked by reading, not compiled, not run on hardware
//
#include <hip/hip_runtime.h>

typedef __attribute__((ext_vector_type(16))) _Float16 v16h;
typedef __attribute__((ext_vector_type(4)))  _Float16 v4h;
typedef __attribute__((ext_vector_type(8)))  float    v8f;
typedef __attribute__((ext_vector_type(4)))  float    v4f;

#define GRID_H 11
#define GRID_W 15
#define HEXDIM 64
#define NHEX   165          // 11*15
#define KDIM   448          // 7*64
#define NOUT   64
#define ROWS_PER_WAVE  32   // two 16-row A tiles per wave
#define ROWS_PER_BLOCK 256  // 8 waves * 32 rows

__global__ __launch_bounds__(256, 2)
void hexconv_wmma_kernel(const float* __restrict__ x,
                         const float* __restrict__ W,
                         const float* __restrict__ bias,
                         float* __restrict__ out)
{
    // W staged as f16 in LDS: sW[n*KDIM + k] = (f16) W[n,k]   (57,344 B)
    __shared__ alignas(32) _Float16 sW[NOUT * KDIM];
    {
        const v4f* W4 = (const v4f*)W;
        for (int i = threadIdx.x; i < (NOUT * KDIM) / 4; i += 256) {
            v4f w = W4[i];
            v4h h;
            h[0] = (_Float16)w[0];
            h[1] = (_Float16)w[1];
            h[2] = (_Float16)w[2];
            h[3] = (_Float16)w[3];
            *(v4h*)(&sW[i * 4]) = h;   // ds_store_b64
        }
    }
    __syncthreads();

    const int wave = threadIdx.x >> 5;   // wave32
    const int lane = threadIdx.x & 31;
    const int hi   = lane >> 4;          // K-half selector (A/B layout)
    const int lm   = lane & 15;          // A: row-in-tile ; B/C: col-in-tile

    const int rowBase = blockIdx.x * ROWS_PER_BLOCK + wave * ROWS_PER_WAVE;

    // ---- A-side: two gather rows per lane (tile 0: rows 0-15, tile 1: rows 16-31)
    const int grow0 = rowBase + lm;
    const int grow1 = rowBase + 16 + lm;

    const int b0i = grow0 / NHEX;
    const int h0  = grow0 - b0i * NHEX;
    const int y0  = h0 / GRID_W;
    const int x0c = h0 - y0 * GRID_W;
    const float* xb0 = x + (size_t)b0i * NHEX * HEXDIM;

    const int b1i = grow1 / NHEX;
    const int h1  = grow1 - b1i * NHEX;
    const int y1  = h1 / GRID_W;
    const int x1c = h1 - y1 * GRID_W;
    const float* xb1 = x + (size_t)b1i * NHEX * HEXDIM;

    // hex neighbor offsets (0-based grid coords). padded-y parity = !(y&1)
    const int dy [7] = {-1,-1, 0, 0, 0, 1, 1};
    const int dxe[7] = { 0, 1,-1, 0, 1, 0, 1};   // padded y even  (y odd)
    const int dxo[7] = {-1, 0,-1, 0, 1,-1, 0};   // padded y odd   (y even)
    const bool yodd0 = (y0 & 1) != 0;
    const bool yodd1 = (y1 & 1) != 0;

    // accumulators pre-loaded with bias (C/D layout: lane -> n = nt*16+lm)
    v8f acc[2][4];
    {
        #pragma unroll
        for (int nt = 0; nt < 4; ++nt) {
            const float bv = bias[nt * 16 + lm];
            #pragma unroll
            for (int r = 0; r < 8; ++r) { acc[0][nt][r] = bv; acc[1][nt][r] = bv; }
        }
    }

    #pragma unroll
    for (int c = 0; c < 14; ++c) {           // K chunks of 32 (never cross a j-block)
        const int j  = c >> 1;               // neighbor index 0..6
        const int kb = (c & 1) * 32;         // k offset inside the 64-wide j-block
        const int klane = kb + 16 * hi;

        // ---- A tile 0 ----
        const int yy0 = y0 + dy[j];
        const int xx0 = x0c + (yodd0 ? dxe[j] : dxo[j]);
        const bool v0 = (yy0 >= 0) && (yy0 < GRID_H) && (xx0 >= 0) && (xx0 < GRID_W);
        v4f f0 = {0.f,0.f,0.f,0.f}, f1 = f0, f2 = f0, f3 = f0;
        if (v0) {
            const float* src = xb0 + (size_t)(yy0 * GRID_W + xx0) * HEXDIM + klane;
            f0 = *(const v4f*)(src);        // K = 16hi+0..3
            f1 = *(const v4f*)(src + 4);    // K = 16hi+4..7
            f2 = *(const v4f*)(src + 8);    // K = 16hi+8..11
            f3 = *(const v4f*)(src + 12);   // K = 16hi+12..15
        }
        // ---- A tile 1 ----
        const int yy1 = y1 + dy[j];
        const int xx1 = x1c + (yodd1 ? dxe[j] : dxo[j]);
        const bool v1 = (yy1 >= 0) && (yy1 < GRID_H) && (xx1 >= 0) && (xx1 < GRID_W);
        v4f g0 = {0.f,0.f,0.f,0.f}, g1 = g0, g2 = g0, g3 = g0;
        if (v1) {
            const float* src = xb1 + (size_t)(yy1 * GRID_W + xx1) * HEXDIM + klane;
            g0 = *(const v4f*)(src);
            g1 = *(const v4f*)(src + 4);
            g2 = *(const v4f*)(src + 8);
            g3 = *(const v4f*)(src + 12);
        }

        v16h a0, a1;
        #pragma unroll
        for (int i = 0; i < 4; ++i) {
            a0[i]      = (_Float16)f0[i];
            a0[4 + i]  = (_Float16)f1[i];
            a0[8 + i]  = (_Float16)f2[i];
            a0[12 + i] = (_Float16)f3[i];
            a1[i]      = (_Float16)g0[i];
            a1[4 + i]  = (_Float16)g1[i];
            a1[8 + i]  = (_Float16)g2[i];
            a1[12 + i] = (_Float16)g3[i];
        }

        // B f16 32x16 layout: lane lm = N-in-tile, halves = K = c*32 + 16*hi + 0..15
        const _Float16* bbase = &sW[(size_t)lm * KDIM + c * 32 + 16 * hi];
        #pragma unroll
        for (int nt = 0; nt < 4; ++nt) {
            const v16h bf = *(const v16h*)(bbase + (size_t)nt * 16 * KDIM);
            acc[0][nt] = __builtin_amdgcn_wmma_f32_16x16x32_f16(
                false, a0, false, bf, (short)0, acc[0][nt], false, false);
            acc[1][nt] = __builtin_amdgcn_wmma_f32_16x16x32_f16(
                false, a1, false, bf, (short)0, acc[1][nt], false, false);
        }
    }

    // C/D layout: VGPR r, lane: m = r + 8*hi (+16 for tile 1), n = nt*16 + lm
    #pragma unroll
    for (int t = 0; t < 2; ++t) {
        float* outRow = out + (size_t)(rowBase + t * 16 + 8 * hi) * NOUT + lm;
        #pragma unroll
        for (int r = 0; r < 8; ++r) {
            float* o = outRow + (size_t)r * NOUT;
            o[0]  = acc[t][0][r];
            o[16] = acc[t][1][r];
            o[32] = acc[t][2][r];
            o[48] = acc[t][3][r];
        }
    }
}

extern "C" void kernel_launch(void* const* d_in, const int* in_sizes, int n_in,
                              void* d_out, int out_size, void* d_ws, size_t ws_size,
                              hipStream_t stream) {
    const float* x    = (const float*)d_in[0];   // (2048, 165, 64) fp32
    const float* W    = (const float*)d_in[1];   // (64, 448) fp32
    const float* bias = (const float*)d_in[2];   // (64,) fp32
    float* out = (float*)d_out;                  // (2048, 165, 64) fp32

    // M_total = 2048*165 = 337920 = 1320 * 256 (exact, no tail)
    dim3 grid(1320), block(256);
    hipLaunchKernelGGL(hexconv_wmma_kernel, grid, block, 0, stream, x, W, bias, out);
    (void)in_sizes; (void)n_in; (void)out_size; (void)d_ws; (void)ws_size;
}
